// SoftRobotTrajectoryPlanner_54975581389041
// MI455X (gfx1250) — compile-verified, matrix operand order checked by reading, not run on hardware
//
#include <hip/hip_runtime.h>
#include <math.h>

// ---------------------------------------------------------------------------
// SoftRobotTrajectoryPlanner forward pass for MI455X (gfx1250, wave32, WMMA).
//
// Precision strategy: all dense GEMMs run as v_wmma_f32_16x16x32_bf16
// (bf16 A/B, f32 accumulate) -- CDNA5 has no deep-K fp32 WMMA. Everything
// else (LayerNorm, softmax, LSTM recurrence, edge scatter) stays fp32.
//
// Every GEMM in this net has M and N multiples of 16 (N=50000=3125*16,
// T=2048), so each 16x16 output tile is owned by exactly one wave32 with
// EXEC all-ones around the WMMA (ISA 7.12 requirement). The K loop is split
// into a branchless aligned main loop (float4 loads + prefetch) and one
// guarded remainder chunk (taken only when K%32 != 0, i.e. K=16 inproj),
// so the fast path has no per-lane divergence.
// ---------------------------------------------------------------------------

#define NNODES 50000
#define NEDGES 800000
#define TSEQ   2048
#define SLOPE  0.2f

typedef __attribute__((ext_vector_type(16))) __bf16 v16bf;
typedef __attribute__((ext_vector_type(8)))  float  v8f;

// Unconditional: load 8 consecutive fp32 (two float4), cvt->bf16.
__device__ __forceinline__ void load8f(const float* __restrict__ p,
                                       v16bf& f, int base) {
  const float4* q = (const float4*)p;
  float4 a = q[0], b = q[1];
  f[base + 0] = (__bf16)a.x; f[base + 1] = (__bf16)a.y;
  f[base + 2] = (__bf16)a.z; f[base + 3] = (__bf16)a.w;
  f[base + 4] = (__bf16)b.x; f[base + 5] = (__bf16)b.y;
  f[base + 6] = (__bf16)b.z; f[base + 7] = (__bf16)b.w;
}

// Guarded tail variant: zero-pads past rem valid elements.
__device__ __forceinline__ void load8g(const float* __restrict__ p, int rem,
                                       v16bf& f, int base) {
#pragma unroll
  for (int j = 0; j < 8; ++j)
    f[base + j] = (__bf16)((j < rem) ? p[j] : 0.0f);
}

// C = act(alpha * A @ op(B) + bias + bias2)
//   A: M x K row-major (lda)
//   BT=true : B is N x K row-major (weight layout / K-matrix for Q@K^T)
//   BT=false: B is K x N row-major (V for att@V)
// One wave32 per 16x16 output tile. ACT: 0=none, 1=relu, 2=gelu(exact).
template <int ACT, bool BT>
__global__ __launch_bounds__(32)
void k_gemm(const float* __restrict__ A, int lda,
            const float* __restrict__ B, int ldb,
            float* __restrict__ C, int ldc,
            const float* __restrict__ bias,
            const float* __restrict__ bias2,
            float alpha, int M, int N, int K) {
  const int lane = threadIdx.x;       // 0..31, one wave
  const int hi   = lane >> 4;         // lane half (ISA fragment layout)
  const int l15  = lane & 15;
  const int mrow = blockIdx.y * 16 + l15;
  const int ncol = blockIdx.x * 16 + l15;

  const float* pa = A + (size_t)mrow * lda + hi * 8;   // A fragment base
  const float* pb = B + (size_t)ncol * ldb + hi * 8;   // NT B fragment base

  const int kAl = K & ~31;            // aligned K extent (wave-uniform)
  v8f acc = {};
  int kk = 0;
#pragma unroll 2
  for (; kk < kAl; kk += 32) {
    // A fragment: 16-bit A 16x32 layout (ISA 7.12.2): lane half 0 holds
    // K=[kk..kk+8) and [kk+16..kk+24); half 1 holds K+8 of each group.
    v16bf af, bfr;
    load8f(pa + kk,      af, 0);
    load8f(pa + kk + 16, af, 8);
    if (BT) {
      load8f(pb + kk,      bfr, 0);
      load8f(pb + kk + 16, bfr, 8);
      __builtin_prefetch(pb + kk + 32, 0, 1);   // global_prefetch_b8
    } else {
#pragma unroll
      for (int j = 0; j < 16; ++j) {
        int k = kk + hi * 8 + ((j < 8) ? j : j + 8);
        bfr[j] = (__bf16)B[(size_t)k * ldb + ncol];
      }
    }
    __builtin_prefetch(pa + kk + 32, 0, 1);     // next A chunk into caches
    acc = __builtin_amdgcn_wmma_f32_16x16x32_bf16(
        /*neg_a=*/false, af, /*neg_b=*/false, bfr,
        /*c_mod=*/(short)0, acc, /*reuse_a=*/false, /*reuse_b=*/false);
  }
  if (kk < K) {                       // only when K%32 != 0 (K=16 inproj)
    const int k0 = kk + hi * 8;
    v16bf af, bfr;
    load8g(pa + kk,      K - k0,        af, 0);
    load8g(pa + kk + 16, K - (k0 + 16), af, 8);
    if (BT) {
      load8g(pb + kk,      K - k0,        bfr, 0);
      load8g(pb + kk + 16, K - (k0 + 16), bfr, 8);
    } else {
#pragma unroll
      for (int j = 0; j < 16; ++j) {
        int k = kk + hi * 8 + ((j < 8) ? j : j + 8);
        bfr[j] = (__bf16)((k < K) ? B[(size_t)k * ldb + ncol] : 0.0f);
      }
    }
    acc = __builtin_amdgcn_wmma_f32_16x16x32_bf16(
        false, af, false, bfr, (short)0, acc, false, false);
  }
  // C/D 16x16 f32 layout: VGPR r -> (m = r + 8*half, n = lane&15)
#pragma unroll
  for (int r = 0; r < 8; ++r) {
    int m = blockIdx.y * 16 + r + hi * 8;
    int n = blockIdx.x * 16 + l15;
    float v = acc[r] * alpha;
    if (bias)  v += bias[n];
    if (bias2) v += bias2[n];
    if (ACT == 1) v = fmaxf(v, 0.0f);
    if (ACT == 2) v = 0.5f * v * (1.0f + erff(v * 0.70710678118f));
    C[(size_t)m * ldc + n] = v;
  }
}

// -------------------------- small fused kernels ----------------------------

__device__ __forceinline__ float blk_sum(float v, float* sh, int bs) {
  sh[threadIdx.x] = v; __syncthreads();
  for (int s = bs >> 1; s > 0; s >>= 1) {
    if ((int)threadIdx.x < s) sh[threadIdx.x] += sh[threadIdx.x + s];
    __syncthreads();
  }
  float r = sh[0]; __syncthreads(); return r;
}

// condition MLP: c = relu(cond@W1^T+b1)@W2^T+b2 ; scale = sig(c)*sig(flex)+0.5
__global__ void k_cond(const float* __restrict__ cond,
                       const float* __restrict__ W1, const float* __restrict__ b1,
                       const float* __restrict__ W2, const float* __restrict__ b2,
                       const float* __restrict__ flex, float* __restrict__ scale) {
  __shared__ float t1[64];
  int j = threadIdx.x;
  float a = b1[j];
  for (int i = 0; i < 4; ++i) a += cond[i] * W1[j * 4 + i];
  t1[j] = fmaxf(a, 0.f);
  __syncthreads();
  float c = b2[j];
  for (int k = 0; k < 64; ++k) c += t1[k] * W2[j * 64 + k];
  float sf = 1.f / (1.f + expf(-flex[0]));
  scale[j] = (1.f / (1.f + expf(-c))) * sf + 0.5f;
}

// input proj epilogue: h = relu(LN(xW+b)) * scale   (block = 64 = features)
__global__ void k_inproj_post(const float* __restrict__ xin,
                              const float* __restrict__ g, const float* __restrict__ b,
                              const float* __restrict__ scale, float* __restrict__ out) {
  __shared__ float sh[64];
  int n = blockIdx.x, f = threadIdx.x;
  float v = xin[(size_t)n * 64 + f];
  float mean = blk_sum(v, sh, 64) * (1.f / 64.f);
  float var  = blk_sum(v * v, sh, 64) * (1.f / 64.f) - mean * mean;
  float y = (v - mean) * rsqrtf(var + 1e-5f) * g[f] + b[f];
  out[(size_t)n * 64 + f] = fmaxf(y, 0.f) * scale[f];
}

// per-node GAT prep: es/ed dots, init segment max/sum, zero accumulator
__global__ void k_gat_node(const float* __restrict__ xh,
                           const float* __restrict__ as, const float* __restrict__ ad,
                           float* __restrict__ es, float* __restrict__ ed,
                           float* __restrict__ mmax, float* __restrict__ ssum,
                           float* __restrict__ accum, int n, int H, int C) {
  int i = blockIdx.x * blockDim.x + threadIdx.x;
  if (i >= n) return;
  for (int h = 0; h < H; ++h) {
    float e1 = 0.f, e2 = 0.f;
    const float* row = xh + (size_t)i * H * C + h * C;
    for (int c = 0; c < C; ++c) { e1 += row[c] * as[h * C + c]; e2 += row[c] * ad[h * C + c]; }
    es[i * H + h] = e1; ed[i * H + h] = e2;
    mmax[i * H + h] = -1e30f; ssum[i * H + h] = 0.f;
  }
  float* arow = accum + (size_t)i * H * C;
  for (int c = 0; c < H * C; ++c) arow[c] = 0.f;
}

__device__ __forceinline__ void edge_sd(int t, const int* __restrict__ ei, int E,
                                        int& s, int& d) {
  if (t < E) { s = ei[t]; d = ei[E + t]; }   // edge_index rows 0/1
  else       { s = t - E; d = s; }           // appended self-loops
}

// sign-aware float atomic max (signed max for +, unsigned min for -)
__device__ __forceinline__ void atomicMaxF(float* a, float v) {
  if (v >= 0.f) atomicMax((int*)a, __float_as_int(v));
  else          atomicMin((unsigned int*)a, __float_as_uint(v));
}

__global__ void k_edge_max(const int* __restrict__ ei, int E, int n,
                           const float* __restrict__ es, const float* __restrict__ ed,
                           float* __restrict__ mmax, int H) {
  int t = blockIdx.x * blockDim.x + threadIdx.x;
  if (t >= E + n) return;
  int s, d; edge_sd(t, ei, E, s, d);
  for (int h = 0; h < H; ++h) {
    float e = es[s * H + h] + ed[d * H + h];
    e = (e > 0.f) ? e : SLOPE * e;
    atomicMaxF(&mmax[d * H + h], e);
  }
}

__global__ void k_edge_sum(const int* __restrict__ ei, int E, int n,
                           const float* __restrict__ es, const float* __restrict__ ed,
                           const float* __restrict__ mmax, float* __restrict__ ssum, int H) {
  int t = blockIdx.x * blockDim.x + threadIdx.x;
  if (t >= E + n) return;
  int s, d; edge_sd(t, ei, E, s, d);
  for (int h = 0; h < H; ++h) {
    float e = es[s * H + h] + ed[d * H + h];
    e = (e > 0.f) ? e : SLOPE * e;
    atomicAdd(&ssum[d * H + h], expf(e - mmax[d * H + h]));
  }
}

__global__ void k_edge_scatter(const int* __restrict__ ei, int E, int n,
                               const float* __restrict__ es, const float* __restrict__ ed,
                               const float* __restrict__ mmax, const float* __restrict__ ssum,
                               const float* __restrict__ xh, float* __restrict__ accum,
                               int H, int C) {
  int t = blockIdx.x * blockDim.x + threadIdx.x;
  if (t >= E + n) return;
  int s, d; edge_sd(t, ei, E, s, d);
  for (int h = 0; h < H; ++h) {
    float e = es[s * H + h] + ed[d * H + h];
    e = (e > 0.f) ? e : SLOPE * e;
    float alpha = expf(e - mmax[d * H + h]) / (ssum[d * H + h] + 1e-16f);
    const float* xrow = xh + (size_t)s * H * C + h * C;
    float* arow = accum + (size_t)d * H * C + h * C;
    for (int c = 0; c < C; ++c) atomicAdd(&arow[c], xrow[c] * alpha);
  }
}

// GAT epilogue: out = [LN(] relu(accum + bias) [)]    (block = F features)
__global__ void k_gat_finish(const float* __restrict__ accum, const float* __restrict__ bias,
                             const float* __restrict__ g, const float* __restrict__ b,
                             float* __restrict__ out, int F, int doLN) {
  __shared__ float sh[128];
  int n = blockIdx.x, f = threadIdx.x;
  float v = fmaxf(accum[(size_t)n * F + f] + bias[f], 0.f);
  if (doLN) {
    float mean = blk_sum(v, sh, F) / (float)F;
    float var  = blk_sum(v * v, sh, F) / (float)F - mean * mean;
    v = (v - mean) * rsqrtf(var + 1e-5f) * g[f] + b[f];
  }
  out[(size_t)n * F + f] = v;
}

__global__ void k_gather(const float* __restrict__ h, const int* __restrict__ idx,
                         float* __restrict__ seq, int T, int F) {
  int t = blockIdx.x * blockDim.x + threadIdx.x;
  if (t >= T * F) return;
  int r = t / F, f = t - r * F;
  seq[t] = h[(size_t)idx[r] * F + f];
}

// One block per direction; 256 threads = one gate lane each. Whh row lives
// in VGPRs; h lives in LDS. pre = x@Wih^T + bih + bhh precomputed via WMMA.
__global__ __launch_bounds__(256)
void k_lstm_scan(const float* __restrict__ preF, const float* __restrict__ preB,
                 const float* __restrict__ WhhF, const float* __restrict__ WhhB,
                 float* __restrict__ out, int ldo, int T) {
  __shared__ float hsh[64];
  __shared__ float gsh[256];
  const int dir = blockIdx.x;
  const float* pre = dir ? preB : preF;
  const float* Whh = dir ? WhhB : WhhF;
  float* ob = out + dir * 64;
  const int j = threadIdx.x;
  float w[64];
#pragma unroll
  for (int i = 0; i < 64; ++i) w[i] = Whh[j * 64 + i];
  float creg = 0.f;
  if (j < 64) hsh[j] = 0.f;
  __syncthreads();
  for (int step = 0; step < T; ++step) {
    const int t = dir ? (T - 1 - step) : step;
    float gsum = pre[(size_t)t * 256 + j];
#pragma unroll
    for (int i = 0; i < 64; ++i) gsum += w[i] * hsh[i];
    gsh[j] = gsum;
    __syncthreads();
    if (j < 64) {               // torch gate order: i, f, g, o
      float ig = 1.f / (1.f + expf(-gsh[j]));
      float fg = 1.f / (1.f + expf(-gsh[64 + j]));
      float gg = tanhf(gsh[128 + j]);
      float og = 1.f / (1.f + expf(-gsh[192 + j]));
      creg = fg * creg + ig * gg;
      float hv = og * tanhf(creg);
      hsh[j] = hv;
      ob[(size_t)t * ldo + j] = hv;
    }
    __syncthreads();
  }
}

__global__ void k_softmax_row(float* __restrict__ S, int W) {  // block 256
  __shared__ float sh[256];
  float* row = S + (size_t)blockIdx.x * W;
  int tid = threadIdx.x;
  float mx = -1e30f;
  for (int i = tid; i < W; i += 256) mx = fmaxf(mx, row[i]);
  sh[tid] = mx; __syncthreads();
  for (int s = 128; s > 0; s >>= 1) { if (tid < s) sh[tid] = fmaxf(sh[tid], sh[tid + s]); __syncthreads(); }
  mx = sh[0]; __syncthreads();
  float sm = 0.f;
  for (int i = tid; i < W; i += 256) { float e = expf(row[i] - mx); row[i] = e; sm += e; }
  sm = blk_sum(sm, sh, 256);
  float inv = 1.f / sm;
  for (int i = tid; i < W; i += 256) row[i] *= inv;
}

__global__ void k_add_ln(float* __restrict__ z, const float* __restrict__ a,
                         const float* __restrict__ g, const float* __restrict__ b) {
  __shared__ float sh[256];
  int t = blockIdx.x, f = threadIdx.x;
  float v = z[(size_t)t * 256 + f] + a[(size_t)t * 256 + f];
  float mean = blk_sum(v, sh, 256) * (1.f / 256.f);
  float var  = blk_sum(v * v, sh, 256) * (1.f / 256.f) - mean * mean;
  z[(size_t)t * 256 + f] = (v - mean) * rsqrtf(var + 1e-5f) * g[f] + b[f];
}

__global__ void k_head_final(const float* __restrict__ h2, const float* __restrict__ W,
                             const float* __restrict__ b, float* __restrict__ out, int T) {
  int t = blockIdx.x * blockDim.x + threadIdx.x;
  if (t >= T) return;
  for (int o = 0; o < 2; ++o) {
    float acc = b[o];
    for (int k = 0; k < 128; ++k) acc += h2[(size_t)t * 128 + k] * W[o * 128 + k];
    out[t * 2 + o] = acc;
  }
}

// ------------------------------- launch ------------------------------------

static inline void gemm(hipStream_t st, int act, bool bt,
                        const float* A, int lda, const float* B, int ldb,
                        float* C, int ldc, const float* bias, const float* bias2,
                        float alpha, int M, int N, int K) {
  dim3 g(N / 16, M / 16), blk(32);
  if (bt) {
    if (act == 0)      k_gemm<0, true><<<g, blk, 0, st>>>(A, lda, B, ldb, C, ldc, bias, bias2, alpha, M, N, K);
    else if (act == 1) k_gemm<1, true><<<g, blk, 0, st>>>(A, lda, B, ldb, C, ldc, bias, bias2, alpha, M, N, K);
    else               k_gemm<2, true><<<g, blk, 0, st>>>(A, lda, B, ldb, C, ldc, bias, bias2, alpha, M, N, K);
  } else {
    k_gemm<0, false><<<g, blk, 0, st>>>(A, lda, B, ldb, C, ldc, bias, bias2, alpha, M, N, K);
  }
}

// workspace layout (floats), 256B aligned blocks; ~116 MB total
constexpr size_t alw(size_t x) { return (x + 63) & ~(size_t)63; }
constexpr size_t O_BUFH  = 0;
constexpr size_t O_BUFXH = O_BUFH  + alw((size_t)NNODES * 128);
constexpr size_t O_BUFG  = O_BUFXH + alw((size_t)NNODES * 128);
constexpr size_t O_ES    = O_BUFG  + alw((size_t)NNODES * 128);
constexpr size_t O_ED    = O_ES    + alw((size_t)NNODES * 2);
constexpr size_t O_MM    = O_ED    + alw((size_t)NNODES * 2);
constexpr size_t O_SS    = O_MM    + alw((size_t)NNODES * 2);
constexpr size_t O_SCALE = O_SS    + alw((size_t)NNODES * 2);
constexpr size_t O_SEQA  = O_SCALE + 64;
constexpr size_t O_SEQB  = O_SEQA  + alw((size_t)TSEQ * 256);
constexpr size_t O_PREF  = O_SEQB  + alw((size_t)TSEQ * 256);
constexpr size_t O_PREB  = O_PREF  + alw((size_t)TSEQ * 256);
constexpr size_t O_QKV   = O_PREB  + alw((size_t)TSEQ * 256);
constexpr size_t O_SC    = O_QKV   + alw((size_t)TSEQ * 768);
constexpr size_t O_ATT   = O_SC    + alw((size_t)TSEQ * TSEQ);
constexpr size_t O_FF    = O_ATT   + alw((size_t)TSEQ * 256);

extern "C" void kernel_launch(void* const* d_in, const int* in_sizes, int n_in,
                              void* d_out, int out_size, void* d_ws, size_t ws_size,
                              hipStream_t stream) {
  // Flattened param order (depth-first insertion order of setup_inputs):
  // 0:x 1:edge_index 2:traj_indices 3:cond 4..12:inp/cond/flex
  // 13..28: g1/n1/g2/n2/g3   29..36: proj + heads
  // 37..60: lstm[3] x {Wih_f,Whh_f,bih_f,bhh_f,Wih_b,Whh_b,bih_b,bhh_b}
  // 61..108: tr[4] x {Wq,bq,Wo,bo,ln1_g,ln1_b,ff1_W,ff1_b,ff2_W,ff2_b,ln2_g,ln2_b}
  auto P = [&](int i) { return (const float*)d_in[i]; };
  const float* x    = P(0);
  const int*   ei   = (const int*)d_in[1];
  const int*   traj = (const int*)d_in[2];
  float* ws = (float*)d_ws;
  float* bufH  = ws + O_BUFH;   float* bufXH = ws + O_BUFXH;
  float* bufG  = ws + O_BUFG;
  float* es = ws + O_ES; float* ed = ws + O_ED;
  float* mm = ws + O_MM; float* ss = ws + O_SS;
  float* scale = ws + O_SCALE;
  float* seqA = ws + O_SEQA; float* seqB = ws + O_SEQB;
  float* preF = ws + O_PREF; float* preB = ws + O_PREB;
  float* qkv  = ws + O_QKV;  float* sc   = ws + O_SC;
  float* att  = ws + O_ATT;  float* ff   = ws + O_FF;

  const int NN = NNODES, E = NEDGES, T = TSEQ;
  const int EE = E + NN;
  const int nodeBlk = (NN + 255) / 256, edgeBlk = (EE + 255) / 256;

  // --- input proj + conditioning --------------------------------------------
  k_cond<<<1, 64, 0, stream>>>(P(3), P(8), P(9), P(10), P(11), P(12), scale);
  gemm(stream, 0, true, x, 16, P(4), 16, bufXH, 64, P(5), nullptr, 1.f, NN, 64, 16);
  k_inproj_post<<<NN, 64, 0, stream>>>(bufXH, P(6), P(7), scale, bufH);

  // --- GAT layers -----------------------------------------------------------
  struct GatCfg { int W, as, ad, b, ng, nb, H, din, dout, doLN; };
  const GatCfg gc[3] = {
    {13, 14, 15, 16, 17, 18, 2,  64, 128, 1},
    {19, 20, 21, 22, 23, 24, 2, 128, 128, 1},
    {25, 26, 27, 28, -1, -1, 1, 128,  64, 0},
  };
  for (int l = 0; l < 3; ++l) {
    const GatCfg& c = gc[l];
    const int C = 64, F = c.dout;
    gemm(stream, 0, true, bufH, c.din, P(c.W), c.din, bufXH, F, nullptr, nullptr, 1.f, NN, F, c.din);
    k_gat_node<<<nodeBlk, 256, 0, stream>>>(bufXH, P(c.as), P(c.ad), es, ed, mm, ss, bufG, NN, c.H, C);
    k_edge_max<<<edgeBlk, 256, 0, stream>>>(ei, E, NN, es, ed, mm, c.H);
    k_edge_sum<<<edgeBlk, 256, 0, stream>>>(ei, E, NN, es, ed, mm, ss, c.H);
    k_edge_scatter<<<edgeBlk, 256, 0, stream>>>(ei, E, NN, es, ed, mm, ss, bufXH, bufG, c.H, C);
    k_gat_finish<<<NN, F, 0, stream>>>(bufG, P(c.b),
                                       c.doLN ? P(c.ng) : nullptr,
                                       c.doLN ? P(c.nb) : nullptr, bufH, F, c.doLN);
  }

  // --- gather trajectory + BiLSTM stack ------------------------------------
  k_gather<<<(T * 64 + 255) / 256, 256, 0, stream>>>(bufH, traj, seqA, T, 64);
  const float* seqIn = seqA; float* seqOut = seqB; int din = 64;
  for (int l = 0; l < 3; ++l) {
    const int b0 = 37 + l * 8;
    gemm(stream, 0, true, seqIn, din, P(b0 + 0), din, preF, 256, P(b0 + 2), P(b0 + 3), 1.f, T, 256, din);
    gemm(stream, 0, true, seqIn, din, P(b0 + 4), din, preB, 256, P(b0 + 6), P(b0 + 7), 1.f, T, 256, din);
    k_lstm_scan<<<2, 256, 0, stream>>>(preF, preB, P(b0 + 1), P(b0 + 5), seqOut, 128, T);
    seqIn = seqOut; seqOut = (seqOut == seqB) ? seqA : seqB; din = 128;
  }
  // after 3 layers: seqIn == seqB (T x 128)

  // --- projection to transformer dim ---------------------------------------
  float* z = seqA;                                   // T x 256
  gemm(stream, 0, true, seqIn, 128, P(29), 128, z, 256, P(30), nullptr, 1.f, T, 256, 128);

  // --- transformer encoder layers ------------------------------------------
  for (int l = 0; l < 4; ++l) {
    const int tb = 61 + l * 12;
    gemm(stream, 0, true, z, 256, P(tb + 0), 256, qkv, 768, P(tb + 1), nullptr, 1.f, T, 768, 256);
    for (int h = 0; h < 4; ++h) {
      const float* Q = qkv + h * 64;
      const float* K = qkv + 256 + h * 64;
      const float* V = qkv + 512 + h * 64;
      gemm(stream, 0, true,  Q, 768, K, 768, sc, T, nullptr, nullptr, 0.125f, T, T, 64);
      k_softmax_row<<<T, 256, 0, stream>>>(sc, T);
      gemm(stream, 0, false, sc, T, V, 768, att + h * 64, 256, nullptr, nullptr, 1.f, T, 64, T);
    }
    gemm(stream, 0, true, att, 256, P(tb + 2), 256, ff, 256, P(tb + 3), nullptr, 1.f, T, 256, 256);
    k_add_ln<<<T, 256, 0, stream>>>(z, ff, P(tb + 4), P(tb + 5));
    gemm(stream, 2, true, z, 256, P(tb + 6), 256, ff, 512, P(tb + 7), nullptr, 1.f, T, 512, 256);
    gemm(stream, 0, true, ff, 512, P(tb + 8), 512, att, 256, P(tb + 9), nullptr, 1.f, T, 256, 512);
    k_add_ln<<<T, 256, 0, stream>>>(z, att, P(tb + 10), P(tb + 11));
  }

  // --- output heads ---------------------------------------------------------
  gemm(stream, 1, true, z, 256, P(31), 256, att, 256, P(32), nullptr, 1.f, T, 256, 256);
  gemm(stream, 1, true, att, 256, P(33), 256, ff, 128, P(34), nullptr, 1.f, T, 128, 256);
  k_head_final<<<(T + 255) / 256, 256, 0, stream>>>(ff, P(35), P(36), (float*)d_out, T);
}
